// RNN_STInf2_41961830482615
// MI455X (gfx1250) — compile-verified, matrix operand order checked by reading, not run on hardware
//
#include <hip/hip_runtime.h>
#include <hip/hip_bf16.h>

// ---------------------------------------------------------------------------
// MI455X (gfx1250) implementation notes
//  - All GEMMs use v_wmma_f32_16x16x32_f16 (f16 operands, f32 accumulate).
//  - Sequential GRU + sampling scan run as persistent 1024-thread blocks
//    (32 wave32s) with hidden state resident in LDS; WMMA per step.
//  - Next-timestep operands are prefetched (global_prefetch_b8) at the top of
//    each sequential step to overlap HBM latency with WMMA work.
//  - LDS leading dims (264 / 104 / 72 halves) chosen so the 16-lane x 16B
//    fragment loads hit distinct LDS banks.
// ---------------------------------------------------------------------------

typedef __attribute__((ext_vector_type(16))) _Float16 v16h;
typedef __attribute__((ext_vector_type(8)))  _Float16 v8h;
typedef __attribute__((ext_vector_type(8)))  float    v8f;

// Dimensions (fixed by the reference)
#define BN   128      // batch
#define TS   512      // T-1
#define HH   256      // hidden
#define G3   768      // 3*H
#define SS   48       // latent
#define DIN  89       // Dd+1+Dt+Db
#define MROW (TS*BN)  // 65536

__device__ __forceinline__ float sigm(float x)  { return 1.0f / (1.0f + __expf(-x)); }
__device__ __forceinline__ float softplusf(float x) { return (x > 20.f) ? x : log1pf(__expf(x)); }

// Load a 16x32 f16 fragment (A or B operand) from a row-major [rows][ld] buffer.
// Lane layout per CDNA5 ISA 7.12.2: lanes 0-15 -> K {0..7,16..23}, lanes 16-31 -> K {8..15,24..31}.
__device__ __forceinline__ v16h load_frag(const _Float16* p, int ld, int row0, int k0) {
  const int lane = threadIdx.x & 31;
  const _Float16* q = p + (size_t)(row0 + (lane & 15)) * ld + (k0 + ((lane >> 4) << 3));
  v8h lo = *(const v8h*)(q);
  v8h hi = *(const v8h*)(q + 16);
  v16h f;
#pragma unroll
  for (int i = 0; i < 8; ++i) { f[i] = lo[i]; f[i + 8] = hi[i]; }
  return f;
}

__device__ __forceinline__ v8f wmma_f16(v16h a, v16h b, v8f c) {
  return __builtin_amdgcn_wmma_f32_16x16x32_f16(false, a, false, b, (short)0, c, false, false);
}

// ---------------------------------------------------------------------------
// Kernel 0: rnn_mask flag per (t,b):  (sum_k Mask[b][t+1][k]) > 1
// ---------------------------------------------------------------------------
__global__ void mask_kernel(const float* __restrict__ Mask, float* __restrict__ flag) {
  int idx = blockIdx.x * 256 + threadIdx.x;          // 0..65535  (t*128+b)
  int t = idx >> 7, b = idx & 127;
  const float* p = Mask + ((size_t)b * 513 + (t + 1)) * 64;
  float s = 0.f;
#pragma unroll 8
  for (int k = 0; k < 64; ++k) s += p[k];
  flag[idx] = (s > 1.f) ? 1.f : 0.f;
}

// ---------------------------------------------------------------------------
// Kernel: f32 -> f16 weight conversion with zero K-padding. dst is [rows][dld].
// ---------------------------------------------------------------------------
__global__ void convert_pad_kernel(_Float16* __restrict__ dst, const float* __restrict__ src,
                                   int rows, int scols, int dld) {
  int idx = blockIdx.x * 256 + threadIdx.x;
  if (idx >= rows * dld) return;
  int r = idx / dld, c = idx % dld;
  dst[idx] = (c < scols) ? (_Float16)src[(size_t)r * scols + c] : (_Float16)0.f;
}

// ---------------------------------------------------------------------------
// Kernel 1: input projections xp = inp @ Wih^T + bih, both directions.
// M=65536 (t*128+b), N=768, K=89 padded to 96.  64x64 block tile, 8 waves.
// ---------------------------------------------------------------------------
__global__ void xp_kernel(const float* __restrict__ x, const float* __restrict__ Trt,
                          const float* __restrict__ Genetic, const float* __restrict__ maskflag,
                          const float* __restrict__ Wih_f, const float* __restrict__ bih_f,
                          const float* __restrict__ Wih_b, const float* __restrict__ bih_b,
                          float* __restrict__ xp_f, float* __restrict__ xp_b) {
  const int dir = blockIdx.z;
  const float* Wih = dir ? Wih_b : Wih_f;
  const float* bih = dir ? bih_b : bih_f;
  float* xp = dir ? xp_b : xp_f;

  __shared__ _Float16 As[64][104];
  __shared__ _Float16 Bs[64][104];
  const int m0 = blockIdx.x * 64;
  const int n0 = blockIdx.y * 64;

  // Stage A: build concat [x | mask | Trt | gen] rows and convert to f16
  for (int i = threadIdx.x; i < 64 * 96; i += 256) {
    int r = i / 96, c = i % 96;
    int gm = m0 + r; int t = gm >> 7; int b = gm & 127;
    float v;
    if (c < 64)       v = x[((size_t)b * 513 + (t + 1)) * 64 + c];
    else if (c == 64) v = maskflag[gm];
    else if (c < 73)  v = Trt[((size_t)b * 513 + (t + 1)) * 8 + (c - 65)];
    else if (c < 89)  v = Genetic[b * 16 + (c - 73)];
    else              v = 0.f;
    As[r][c] = (_Float16)v;
  }
  // Stage B: Wih rows [n0..n0+63], K padded to 96
  for (int i = threadIdx.x; i < 64 * 96; i += 256) {
    int r = i / 96, c = i % 96;
    Bs[r][c] = (c < DIN) ? (_Float16)Wih[(size_t)(n0 + r) * DIN + c] : (_Float16)0.f;
  }
  __syncthreads();

  const int wave = threadIdx.x >> 5;
  const int lane = threadIdx.x & 31;
  for (int tile = wave * 2; tile < wave * 2 + 2; ++tile) {
    int mt = tile >> 2, nt = tile & 3;
    v8f acc = {};
#pragma unroll
    for (int k0 = 0; k0 < 96; k0 += 32) {
      v16h a = load_frag(&As[0][0], 104, mt * 16, k0);
      v16h b = load_frag(&Bs[0][0], 104, nt * 16, k0);
      acc = wmma_f16(a, b, acc);
    }
    int n  = n0 + nt * 16 + (lane & 15);
    int mb = m0 + mt * 16 + ((lane >> 4) << 3);
    float bias = bih[n];
#pragma unroll
    for (int i = 0; i < 8; ++i)
      xp[(size_t)(mb + i) * G3 + n] = acc[i] + bias;
  }
}

// ---------------------------------------------------------------------------
// Kernel 1c: hid_base = [x[:,0] | Genetic | Trt[:,0]] @ W_base^T + b_base
// M=128, N=256, K=88 padded to 96. Single block, 8 waves.
// ---------------------------------------------------------------------------
__global__ void base_kernel(const float* __restrict__ x, const float* __restrict__ Genetic,
                            const float* __restrict__ Trt, const _Float16* __restrict__ Wbase16,
                            const float* __restrict__ b_base, float* __restrict__ hid_base) {
  __shared__ _Float16 As[128][104];
  for (int i = threadIdx.x; i < 128 * 96; i += 256) {
    int b = i / 96, c = i % 96;
    float v;
    if (c < 64)      v = x[((size_t)b * 513) * 64 + c];
    else if (c < 80) v = Genetic[b * 16 + (c - 64)];
    else if (c < 88) v = Trt[((size_t)b * 513) * 8 + (c - 80)];
    else             v = 0.f;
    As[b][c] = (_Float16)v;
  }
  __syncthreads();
  const int wave = threadIdx.x >> 5;
  const int lane = threadIdx.x & 31;
  for (int tile = wave; tile < 128; tile += 8) {
    int mt = tile >> 4, nt = tile & 15;
    v8f acc = {};
#pragma unroll
    for (int k0 = 0; k0 < 96; k0 += 32) {
      v16h a = load_frag(&As[0][0], 104, mt * 16, k0);
      v16h b = load_frag(Wbase16, 96, nt * 16, k0);
      acc = wmma_f16(a, b, acc);
    }
    int n  = nt * 16 + (lane & 15);
    int mb = mt * 16 + ((lane >> 4) << 3);
    float bias = b_base[n];
#pragma unroll
    for (int i = 0; i < 8; ++i)
      hid_base[(size_t)(mb + i) * HH + n] = acc[i] + bias;
  }
}

// ---------------------------------------------------------------------------
// Kernel 2: persistent GRU, one 1024-thread block per direction.
// Per step: gh = h @ Whh^T (+bhh), gates, h update. h double-buffered in LDS.
// Next step's xp block is prefetched to overlap HBM latency with WMMA work.
// Dynamic LDS: 4 * 128*264 f16 buffers (h0, h1, r, z) = 270336 bytes.
// ---------------------------------------------------------------------------
__global__ void gru_kernel(const float* __restrict__ xp_f, const float* __restrict__ xp_b,
                           const _Float16* __restrict__ Whh16_f, const _Float16* __restrict__ Whh16_b,
                           const float* __restrict__ bhh_f, const float* __restrict__ bhh_b,
                           _Float16* __restrict__ hcat) {
  extern __shared__ char smem_raw[];
  _Float16* sm = (_Float16*)smem_raw;
  const int HLD = 264;
  _Float16* hbuf0 = sm;
  _Float16* hbuf1 = sm + 128 * HLD;
  _Float16* rbuf  = sm + 2 * 128 * HLD;
  _Float16* zbuf  = sm + 3 * 128 * HLD;

  const int dir = blockIdx.x;
  const float*    xp  = dir ? xp_b : xp_f;
  const _Float16* W   = dir ? Whh16_b : Whh16_f;
  const float*    bhh = dir ? bhh_b : bhh_f;

  for (int i = threadIdx.x; i < 128 * HLD; i += 1024) hbuf0[i] = (_Float16)0.f;
  __syncthreads();

  const int wave = threadIdx.x >> 5;
  const int lane = threadIdx.x & 31;
  _Float16* hcur = hbuf0;
  _Float16* hnxt = hbuf1;

  for (int s = 0; s < TS; ++s) {
    const int t = dir ? (TS - 1 - s) : s;
    const float* xpt = xp + (size_t)t * (BN * G3);

    // Prefetch next step's xp block (384KB = 3072 x 128B cachelines) so the
    // HBM latency overlaps with this step's WMMA work. global_prefetch_b8.
    if (s + 1 < TS) {
      const int tn = dir ? (t - 1) : (t + 1);
      const char* nxt = (const char*)(xp + (size_t)tn * (BN * G3));
#pragma unroll
      for (int l = 0; l < 3; ++l)
        __builtin_prefetch(nxt + ((size_t)(threadIdx.x * 3 + l)) * 128, 0, 0);
    }

    // Phase A: r/z gates (j tiles 0..31), 256 tiles over 32 waves
    for (int tile = wave; tile < 256; tile += 32) {
      int bt = tile >> 5, jt = tile & 31;
      v8f acc = {};
#pragma unroll
      for (int k0 = 0; k0 < HH; k0 += 32) {
        v16h a = load_frag(hcur, HLD, bt * 16, k0);
        v16h b = load_frag(W, HH, jt * 16, k0);
        acc = wmma_f16(a, b, acc);
      }
      int j  = jt * 16 + (lane & 15);
      int bb = bt * 16 + ((lane >> 4) << 3);
      float bj = bhh[j];
#pragma unroll
      for (int i = 0; i < 8; ++i) {
        int b_ = bb + i;
        float g = acc[i] + bj + xpt[(size_t)b_ * G3 + j];
        float sg = sigm(g);
        if (j < HH) rbuf[b_ * HLD + j]        = (_Float16)sg;
        else        zbuf[b_ * HLD + (j - HH)] = (_Float16)sg;
      }
    }
    __syncthreads();

    // Phase B: n gate (j tiles 32..47) + hidden update, 128 tiles
    for (int tile = wave; tile < 128; tile += 32) {
      int bt = tile >> 4, jt = 32 + (tile & 15);
      v8f acc = {};
#pragma unroll
      for (int k0 = 0; k0 < HH; k0 += 32) {
        v16h a = load_frag(hcur, HLD, bt * 16, k0);
        v16h b = load_frag(W, HH, jt * 16, k0);
        acc = wmma_f16(a, b, acc);
      }
      int j  = jt * 16 + (lane & 15);
      int k  = j - 2 * HH;
      int bb = bt * 16 + ((lane >> 4) << 3);
      float bj = bhh[j];
#pragma unroll
      for (int i = 0; i < 8; ++i) {
        int b_ = bb + i;
        float gn  = acc[i] + bj;
        float r   = (float)rbuf[b_ * HLD + k];
        float z   = (float)zbuf[b_ * HLD + k];
        float n   = tanhf(xpt[(size_t)b_ * G3 + j] + r * gn);
        float h   = (float)hcur[b_ * HLD + k];
        float hn  = (1.f - z) * n + z * h;
        hnxt[b_ * HLD + k] = (_Float16)hn;
        hcat[((size_t)t * BN + b_) * (2 * HH) + dir * HH + k] = (_Float16)hn;
      }
    }
    __syncthreads();
    _Float16* tmp = hcur; hcur = hnxt; hnxt = tmp;
  }
}

// ---------------------------------------------------------------------------
// Kernel 3a: h2 = tanh(hcat @ W_hr^T + b_hr).  M=65536, N=256, K=512.
// ---------------------------------------------------------------------------
__global__ void h2_kernel(const _Float16* __restrict__ hcat, const _Float16* __restrict__ Whr16,
                          const float* __restrict__ b_hr, _Float16* __restrict__ h2) {
  const int m0 = blockIdx.x * 64;
  const int n0 = blockIdx.y * 64;
  const int wave = threadIdx.x >> 5;
  const int lane = threadIdx.x & 31;
  for (int tile = wave * 2; tile < wave * 2 + 2; ++tile) {
    int mt = tile >> 2, nt = tile & 3;
    v8f acc = {};
#pragma unroll
    for (int k0 = 0; k0 < 512; k0 += 32) {
      v16h a = load_frag(hcat, 512, m0 + mt * 16, k0);
      v16h b = load_frag(Whr16, 512, n0 + nt * 16, k0);
      acc = wmma_f16(a, b, acc);
    }
    int n  = n0 + nt * 16 + (lane & 15);
    int mb = m0 + mt * 16 + ((lane >> 4) << 3);
    float bias = b_hr[n];
#pragma unroll
    for (int i = 0; i < 8; ++i)
      h2[(size_t)(mb + i) * HH + n] = (_Float16)tanhf(acc[i] + bias);
  }
}

// ---------------------------------------------------------------------------
// Kernel 3b: mu2 = h2@W_mu2^T + b_mu2 ; sig2 = softplus(h2@W_sig2^T + b_sig2)
// Weights stacked [96][256] (rows 0..47 = W_mu2, 48..95 = W_sig2).
// ---------------------------------------------------------------------------
__global__ void mu2_kernel(const _Float16* __restrict__ h2, const _Float16* __restrict__ Wms16,
                           const float* __restrict__ b_mu2, const float* __restrict__ b_sig2,
                           float* __restrict__ mu2, float* __restrict__ sig2) {
  const int m0 = blockIdx.x * 64;
  const int wave = threadIdx.x >> 5;
  const int lane = threadIdx.x & 31;
  for (int tile = wave * 3; tile < wave * 3 + 3; ++tile) {
    int mt = tile / 6, nt = tile % 6;
    v8f acc = {};
#pragma unroll
    for (int k0 = 0; k0 < HH; k0 += 32) {
      v16h a = load_frag(h2, HH, m0 + mt * 16, k0);
      v16h b = load_frag(Wms16, HH, nt * 16, k0);
      acc = wmma_f16(a, b, acc);
    }
    int n  = nt * 16 + (lane & 15);
    int mb = m0 + mt * 16 + ((lane >> 4) << 3);
#pragma unroll
    for (int i = 0; i < 8; ++i) {
      int m = mb + i;                       // m = t*128 + b
      if (n < SS) mu2[(size_t)m * SS + n]        = acc[i] + b_mu2[n];
      else        sig2[(size_t)m * SS + (n - SS)] = softplusf(acc[i] + b_sig2[n - SS]);
    }
  }
}

// ---------------------------------------------------------------------------
// Kernel 4: sequential sampling scan (product of Gaussians + rsample).
// One persistent 1024-thread block.  Per step:
//   ph1: [mu1|s1] = h1 @ [W_mu|W_sig]^T   (M=128,N=96,K=256)
//   pog: combine with mu2/sig2/eps -> z, write outputs
//   ph2: h1 = tanh(z @ W_hz1^T + b_hz1)   (M=128,N=256,K=64 padded)
// Next step's mu2/sig2/eps rows are prefetched at the top of each step.
// Dynamic LDS: h1[128*264]f16 + z[128*72]f16 + mu1/s1[128*48]f32 each.
// ---------------------------------------------------------------------------
__global__ void scan_kernel(const float* __restrict__ hid_base,
                            const _Float16* __restrict__ Wms16,   // [96][256] W_mu|W_sig
                            const float* __restrict__ b_mu, const float* __restrict__ b_sig,
                            const _Float16* __restrict__ Whz116,  // [256][64]
                            const float* __restrict__ b_hz1,
                            const float* __restrict__ mu2, const float* __restrict__ sig2,
                            const float* __restrict__ eps, float* __restrict__ out) {
  extern __shared__ char smem_raw[];
  const int HLD = 264, ZLD = 72;
  _Float16* h1 = (_Float16*)smem_raw;
  _Float16* zb = h1 + 128 * HLD;
  float* mu1 = (float*)(zb + 128 * ZLD);
  float* s1  = mu1 + 128 * SS;

  for (int i = threadIdx.x; i < 128 * HH; i += 1024) {
    int b = i >> 8, c = i & 255;
    h1[b * HLD + c] = (_Float16)tanhf(hid_base[i]);
  }
  for (int i = threadIdx.x; i < 128 * ZLD; i += 1024) zb[i] = (_Float16)0.f;
  __syncthreads();

  const int wave = threadIdx.x >> 5;
  const int lane = threadIdx.x & 31;

  for (int t = 0; t < TS; ++t) {
    // Prefetch next step's mu2/sig2/eps rows (3 x 24KB = 576 cachelines).
    if (t + 1 < TS) {
      size_t gn = (size_t)(t + 1) * (128 * SS);
      int cl = threadIdx.x;                 // one 128B line per thread, 192 lines each
      if (cl < 192) {
        __builtin_prefetch((const char*)(mu2 + gn) + (size_t)cl * 128, 0, 0);
      } else if (cl < 384) {
        __builtin_prefetch((const char*)(sig2 + gn) + (size_t)(cl - 192) * 128, 0, 0);
      } else if (cl < 576) {
        __builtin_prefetch((const char*)(eps + gn) + (size_t)(cl - 384) * 128, 0, 0);
      }
    }

    // phase 1: mu1/s1
    for (int tile = wave; tile < 48; tile += 32) {
      int mt = tile / 6, nt = tile % 6;
      v8f acc = {};
#pragma unroll
      for (int k0 = 0; k0 < HH; k0 += 32) {
        v16h a = load_frag(h1, HLD, mt * 16, k0);
        v16h b = load_frag(Wms16, HH, nt * 16, k0);
        acc = wmma_f16(a, b, acc);
      }
      int n  = nt * 16 + (lane & 15);
      int mb = mt * 16 + ((lane >> 4) << 3);
#pragma unroll
      for (int i = 0; i < 8; ++i) {
        int b_ = mb + i;
        if (n < SS) mu1[b_ * SS + n]        = acc[i] + b_mu[n];
        else        s1[b_ * SS + (n - SS)]  = softplusf(acc[i] + b_sig[n - SS]);
      }
    }
    __syncthreads();

    // phase pog: combine + rsample, write [3,B,TS,S]
    for (int i = threadIdx.x; i < 128 * SS; i += 1024) {
      int b = i / SS, s = i % SS;
      size_t g = (size_t)t * (128 * SS) + i;      // (t*128+b)*48+s
      float m1 = mu1[i], ss1 = s1[i];
      float m2 = mu2[g], ss2 = sig2[g];
      float v1 = ss1 * ss1 + 1e-8f, v2 = ss2 * ss2 + 1e-8f;
      float v  = v1 * v2 / (v1 + v2);
      float mu = (m1 / v1 + m2 / v2) * v;
      float sg = sqrtf(v);
      float z  = mu + sg * eps[g];
      size_t ob = ((size_t)b * TS + t) * SS + s;
      out[ob]                              = z;
      out[(size_t)BN * TS * SS + ob]       = mu;
      out[(size_t)2 * BN * TS * SS + ob]   = sg;
      zb[b * ZLD + s] = (_Float16)z;
    }
    __syncthreads();

    // phase 2: next h1
    if (t < TS - 1) {
      for (int tile = wave; tile < 128; tile += 32) {
        int mt = tile >> 4, nt = tile & 15;
        v8f acc = {};
#pragma unroll
        for (int k0 = 0; k0 < 64; k0 += 32) {
          v16h a = load_frag(zb, ZLD, mt * 16, k0);
          v16h b = load_frag(Whz116, 64, nt * 16, k0);
          acc = wmma_f16(a, b, acc);
        }
        int n  = nt * 16 + (lane & 15);
        int mb = mt * 16 + ((lane >> 4) << 3);
#pragma unroll
        for (int i = 0; i < 8; ++i)
          h1[(mb + i) * HLD + n] = (_Float16)tanhf(acc[i] + b_hz1[n]);
      }
      __syncthreads();
    }
  }
}

// ---------------------------------------------------------------------------
// Host launch
// ---------------------------------------------------------------------------
extern "C" void kernel_launch(void* const* d_in, const int* in_sizes, int n_in,
                              void* d_out, int out_size, void* d_ws, size_t ws_size,
                              hipStream_t stream) {
  (void)in_sizes; (void)n_in; (void)out_size; (void)ws_size;
  const float* x       = (const float*)d_in[0];
  const float* Trt     = (const float*)d_in[1];
  const float* Mask    = (const float*)d_in[2];
  const float* Genetic = (const float*)d_in[3];
  const float* eps     = (const float*)d_in[4];
  const float* Wih_f   = (const float*)d_in[5];
  const float* Whh_f   = (const float*)d_in[6];
  const float* bih_f   = (const float*)d_in[7];
  const float* bhh_f   = (const float*)d_in[8];
  const float* Wih_b   = (const float*)d_in[9];
  const float* Whh_b   = (const float*)d_in[10];
  const float* bih_b   = (const float*)d_in[11];
  const float* bhh_b   = (const float*)d_in[12];
  const float* W_base  = (const float*)d_in[13];
  const float* b_base  = (const float*)d_in[14];
  const float* W_hr    = (const float*)d_in[15];
  const float* b_hr    = (const float*)d_in[16];
  const float* W_hz1   = (const float*)d_in[17];
  const float* b_hz1   = (const float*)d_in[18];
  const float* W_mu    = (const float*)d_in[19];
  const float* b_mu    = (const float*)d_in[20];
  const float* W_sig   = (const float*)d_in[21];
  const float* b_sig   = (const float*)d_in[22];
  const float* W_mu2   = (const float*)d_in[23];
  const float* b_mu2   = (const float*)d_in[24];
  const float* W_sig2  = (const float*)d_in[25];
  const float* b_sig2  = (const float*)d_in[26];
  float* out = (float*)d_out;

  // Workspace carve-out (~530 MB total)
  char* ws = (char*)d_ws;
  auto alloc = [&](size_t bytes) -> char* {
    char* p = ws; ws += (bytes + 255) & ~(size_t)255; return p;
  };
  float*    maskflag = (float*)alloc((size_t)MROW * 4);
  float*    xp_f     = (float*)alloc((size_t)MROW * G3 * 4);
  float*    xp_b     = (float*)alloc((size_t)MROW * G3 * 4);
  _Float16* Whh16_f  = (_Float16*)alloc((size_t)G3 * HH * 2);
  _Float16* Whh16_b  = (_Float16*)alloc((size_t)G3 * HH * 2);
  _Float16* Whr16    = (_Float16*)alloc((size_t)HH * 512 * 2);
  _Float16* Wms2_16  = (_Float16*)alloc((size_t)96 * HH * 2);   // W_mu2|W_sig2
  _Float16* Wms_16   = (_Float16*)alloc((size_t)96 * HH * 2);   // W_mu|W_sig
  _Float16* Whz116   = (_Float16*)alloc((size_t)HH * 64 * 2);
  _Float16* Wbase16  = (_Float16*)alloc((size_t)HH * 96 * 2);
  float*    hid_base = (float*)alloc((size_t)BN * HH * 4);
  _Float16* hcat     = (_Float16*)alloc((size_t)MROW * 512 * 2);
  _Float16* h2       = (_Float16*)alloc((size_t)MROW * HH * 2);
  float*    mu2      = (float*)alloc((size_t)MROW * SS * 4);
  float*    sig2     = (float*)alloc((size_t)MROW * SS * 4);

  mask_kernel<<<MROW / 256, 256, 0, stream>>>(Mask, maskflag);

  auto cvt = [&](_Float16* d, const float* s, int r, int sc, int dl) {
    int total = r * dl;
    convert_pad_kernel<<<(total + 255) / 256, 256, 0, stream>>>(d, s, r, sc, dl);
  };
  cvt(Whh16_f, Whh_f, G3, HH, HH);
  cvt(Whh16_b, Whh_b, G3, HH, HH);
  cvt(Whr16, W_hr, HH, 512, 512);
  cvt(Wms2_16,            W_mu2,  SS, HH, HH);
  cvt(Wms2_16 + SS * HH,  W_sig2, SS, HH, HH);
  cvt(Wms_16,             W_mu,   SS, HH, HH);
  cvt(Wms_16 + SS * HH,   W_sig,  SS, HH, HH);
  cvt(Whz116, W_hz1, HH, SS, 64);
  cvt(Wbase16, W_base, HH, 88, 96);

  dim3 g1(MROW / 64, G3 / 64, 2);
  xp_kernel<<<g1, 256, 0, stream>>>(x, Trt, Genetic, maskflag,
                                    Wih_f, bih_f, Wih_b, bih_b, xp_f, xp_b);
  base_kernel<<<1, 256, 0, stream>>>(x, Genetic, Trt, Wbase16, b_base, hid_base);

  size_t gru_lds = (size_t)4 * 128 * 264 * sizeof(_Float16);   // 270336 B
  gru_kernel<<<2, 1024, gru_lds, stream>>>(xp_f, xp_b, Whh16_f, Whh16_b,
                                           bhh_f, bhh_b, hcat);

  h2_kernel<<<dim3(MROW / 64, HH / 64), 256, 0, stream>>>(hcat, Whr16, b_hr, h2);
  mu2_kernel<<<MROW / 64, 256, 0, stream>>>(h2, Wms2_16, b_mu2, b_sig2, mu2, sig2);

  size_t scan_lds = (size_t)128 * 264 * 2 + (size_t)128 * 72 * 2
                  + (size_t)2 * 128 * SS * 4;                  // 135168 B
  scan_kernel<<<1, 1024, scan_lds, stream>>>(hid_base, Wms_16, b_mu, b_sig,
                                             Whz116, b_hz1, mu2, sig2, eps, out);
}
